// GINLayer_35914516529218
// MI455X (gfx1250) — matrix operand order repaired and checked by reading, not compiled, not run on hardware
//
#include <hip/hip_runtime.h>
#include <hip/hip_bf16.h>

// ---- problem constants (from reference) ----
static constexpr int D       = 128;
static constexpr int NNODES  = 50000;
static constexpr int NEDGES  = 800000;
#define BN_EPS 1e-5f

typedef __bf16 bf16_t;
typedef __attribute__((ext_vector_type(16))) __bf16 v16bf;
typedef __attribute__((ext_vector_type(8)))  __bf16 v8bf;
typedef __attribute__((ext_vector_type(8)))  float  v8f;

// round-to-nearest-even f32 -> bf16 (pure bit ops, no cvt dependency)
__device__ __forceinline__ bf16_t f2bf(float f) {
  unsigned u = __builtin_bit_cast(unsigned, f);
  unsigned r = (u + 0x7FFFu + ((u >> 16) & 1u)) >> 16;
  return __builtin_bit_cast(bf16_t, (unsigned short)r);
}

// hardware f32 atomic add to global memory (avoid CAS-loop expansion)
__device__ __forceinline__ void gatomic_fadd(float* p, float v) {
#if defined(__HIP_DEVICE_COMPILE__)
  unsafeAtomicAdd(p, v);      // lowers to global_atomic_add_f32
#else
  atomicAdd(p, v);
#endif
}

// -------------------------------------------------------------------------
// k_prep: pack W1/W2 into WMMA-B-fragment order (bf16) + zero BN accumulators.
// Packed layout: tile (nt,kt) -> tileIdx = nt*4+kt; per tile 32 lanes x 16 bf16.
//   lane<16 : element j = W[(kt*32 + j + (j>=8 ? 8 : 0))     * D + nt*16 + lane]
//   lane>=16: element j = W[(kt*32 + 8 + j + (j>=8 ? 8 : 0)) * D + nt*16 + lane-16]
// => GEMM B-load is one contiguous 32B v16bf per lane.
// -------------------------------------------------------------------------
__global__ void k_prep(const float* __restrict__ W1, const float* __restrict__ W2,
                       bf16_t* __restrict__ w1p, bf16_t* __restrict__ w2p,
                       float* __restrict__ gacc /* 4*D floats */) {
  int gid = blockIdx.x * blockDim.x + threadIdx.x;
  if (gid < 4 * D) gacc[gid] = 0.0f;          // gsum, gsq, bnscale, bnshift
  int o = gid;
  const float* W = W1;
  bf16_t*      P = w1p;
  if (o >= 16384) { o -= 16384; W = W2; P = w2p; }
  if (gid < 32768) {
    int tile = o >> 9;          // /512
    int rem  = o & 511;
    int lane = rem >> 4;
    int j    = rem & 15;
    int nt = tile >> 2, kt = tile & 3;
    int n = nt * 16 + (lane & 15);
    int k = kt * 32 + ((lane & 16) ? 8 : 0) + j + ((j & 8) ? 8 : 0);
    P[o] = f2bf(W[k * D + n]);
  }
}

// z0 = (1 + eps) * h
__global__ void k_init(const float* __restrict__ h, const float* __restrict__ eps,
                       float* __restrict__ z0) {
  int i = blockIdx.x * blockDim.x + threadIdx.x;   // over N*D/4
  if (i >= NNODES * D / 4) return;
  float g = 1.0f + eps[0];
  float4 hv = ((const float4*)h)[i];
  ((float4*)z0)[i] = make_float4(hv.x * g, hv.y * g, hv.z * g, hv.w * g);
}

// z0[dst[e]] += h[src[e]] * mask[e]  (32 threads per edge, float4 per thread)
__global__ void k_scatter(const float* __restrict__ h, const float* __restrict__ mask,
                          const int* __restrict__ src, const int* __restrict__ dst,
                          float* __restrict__ z0) {
  long long t = (long long)blockIdx.x * blockDim.x + threadIdx.x;
  int e = (int)(t >> 5);
  int q = (int)(t & 31);
  if (e >= NEDGES) return;
  int s  = src[e];
  int d0 = dst[e];
  float w = mask[e];
  float4 hv = *(const float4*)(h + (size_t)s * D + q * 4);
  float* zp = z0 + (size_t)d0 * D + q * 4;
  gatomic_fadd(zp + 0, hv.x * w);
  gatomic_fadd(zp + 1, hv.y * w);
  gatomic_fadd(zp + 2, hv.z * w);
  gatomic_fadd(zp + 3, hv.w * w);
}

// -------------------------------------------------------------------------
// GEMM1: y = relu(z0 @ W1 + b1), output bf16 row-major.
// One wave = 16 rows x 128 cols; 4 k-steps x 8 n-tiles = 32 WMMA.
// -------------------------------------------------------------------------
__global__ void __launch_bounds__(256) k_gemm1(const float* __restrict__ z0,
                                               const bf16_t* __restrict__ w1p,
                                               const float* __restrict__ b1,
                                               bf16_t* __restrict__ y) {
  const int lane = threadIdx.x & 31;
  const int wave = threadIdx.x >> 5;
  const int m0 = (blockIdx.x * 8 + wave) * 16;
  if (m0 >= NNODES) return;                 // wave-uniform -> EXEC stays full
  const int col = lane & 15;
  const int aHo = (lane >> 4) * 8;          // A-layout K half-offset

  v8f acc[8];
#pragma unroll
  for (int nt = 0; nt < 8; ++nt) {
    float bv = b1[nt * 16 + col];
#pragma unroll
    for (int i = 0; i < 8; ++i) acc[nt][i] = bv;
  }

  const float* ar = z0 + (size_t)(m0 + col) * D;
  const v16bf* bp = (const v16bf*)w1p;

#pragma unroll
  for (int kt = 0; kt < 4; ++kt) {
    const float4* c0 = (const float4*)(ar + kt * 32 + aHo);
    const float4* c1 = (const float4*)(ar + kt * 32 + aHo + 16);
    float4 f0 = c0[0], f1 = c0[1], g0 = c1[0], g1 = c1[1];
    v16bf a;
    a[0]  = f2bf(f0.x); a[1]  = f2bf(f0.y); a[2]  = f2bf(f0.z); a[3]  = f2bf(f0.w);
    a[4]  = f2bf(f1.x); a[5]  = f2bf(f1.y); a[6]  = f2bf(f1.z); a[7]  = f2bf(f1.w);
    a[8]  = f2bf(g0.x); a[9]  = f2bf(g0.y); a[10] = f2bf(g0.z); a[11] = f2bf(g0.w);
    a[12] = f2bf(g1.x); a[13] = f2bf(g1.y); a[14] = f2bf(g1.z); a[15] = f2bf(g1.w);
#pragma unroll
    for (int nt = 0; nt < 8; ++nt) {
      v16bf b = bp[(nt * 4 + kt) * 32 + lane];
      acc[nt] = __builtin_amdgcn_wmma_f32_16x16x32_bf16(
          false, a, false, b, (short)0, acc[nt], false, false);
    }
  }

#pragma unroll
  for (int nt = 0; nt < 8; ++nt) {
    int n = nt * 16 + col;
#pragma unroll
    for (int v = 0; v < 8; ++v) {
      int m = m0 + v + aHo;                 // C layout: lane<16 M=v, lane>=16 M=v+8
      float r = acc[nt][v];
      y[(size_t)m * D + n] = f2bf(r > 0.0f ? r : 0.0f);
    }
  }
}

// -------------------------------------------------------------------------
// GEMM2: t = (y @ W2 + b2) * snorm_n ; write t to d_out (pre-BN buffer),
// and reduce per-column sum / sum-of-squares via LDS then global atomics.
// -------------------------------------------------------------------------
__global__ void __launch_bounds__(256) k_gemm2(const bf16_t* __restrict__ y,
                                               const bf16_t* __restrict__ w2p,
                                               const float* __restrict__ b2,
                                               const float* __restrict__ snorm,
                                               float* __restrict__ outt,
                                               float* __restrict__ gsum,
                                               float* __restrict__ gsq) {
  __shared__ float lsum[D];
  __shared__ float lsq[D];
  if (threadIdx.x < D) { lsum[threadIdx.x] = 0.0f; lsq[threadIdx.x] = 0.0f; }
  __syncthreads();

  const int lane = threadIdx.x & 31;
  const int wave = threadIdx.x >> 5;
  const int m0 = (blockIdx.x * 8 + wave) * 16;
  const int col = lane & 15;
  const int aHo = (lane >> 4) * 8;

  if (m0 < NNODES) {                        // wave-uniform
    v8f acc[8];
#pragma unroll
    for (int nt = 0; nt < 8; ++nt) {
      float bv = b2[nt * 16 + col];
#pragma unroll
      for (int i = 0; i < 8; ++i) acc[nt][i] = bv;
    }

    const bf16_t* ar = y + (size_t)(m0 + col) * D;
    const v16bf* bp = (const v16bf*)w2p;

#pragma unroll
    for (int kt = 0; kt < 4; ++kt) {
      v8bf lo = *(const v8bf*)(ar + kt * 32 + aHo);
      v8bf hi = *(const v8bf*)(ar + kt * 32 + aHo + 16);
      v16bf a;
#pragma unroll
      for (int i = 0; i < 8; ++i) { a[i] = lo[i]; a[8 + i] = hi[i]; }
#pragma unroll
      for (int nt = 0; nt < 8; ++nt) {
        v16bf b = bp[(nt * 4 + kt) * 32 + lane];
        acc[nt] = __builtin_amdgcn_wmma_f32_16x16x32_bf16(
            false, a, false, b, (short)0, acc[nt], false, false);
      }
    }

    float sv[8];
#pragma unroll
    for (int v = 0; v < 8; ++v) sv[v] = snorm[m0 + v + aHo];

#pragma unroll
    for (int nt = 0; nt < 8; ++nt) {
      int n = nt * 16 + col;
      float cs = 0.0f, cq = 0.0f;
#pragma unroll
      for (int v = 0; v < 8; ++v) {
        float val = acc[nt][v] * sv[v];
        outt[(size_t)(m0 + v + aHo) * D + n] = val;
        cs += val;
        cq += val * val;
      }
      atomicAdd(&lsum[n], cs);              // ds_add_f32
      atomicAdd(&lsq[n], cq);
    }
  }
  __syncthreads();
  if (threadIdx.x < D) {
    gatomic_fadd(&gsum[threadIdx.x], lsum[threadIdx.x]);
    gatomic_fadd(&gsq[threadIdx.x],  lsq[threadIdx.x]);
  }
}

// BN statistics -> affine scale/shift
__global__ void k_bnfin(const float* __restrict__ gsum, const float* __restrict__ gsq,
                        const float* __restrict__ gamma, const float* __restrict__ beta,
                        float* __restrict__ bnscale, float* __restrict__ bnshift) {
  int d = threadIdx.x;
  if (d < D) {
    float invN = 1.0f / (float)NNODES;
    float mu  = gsum[d] * invN;
    float var = gsq[d] * invN - mu * mu;
    float sc  = gamma[d] * rsqrtf(var + BN_EPS);
    bnscale[d] = sc;
    bnshift[d] = beta[d] - mu * sc;
  }
}

// out = h + relu(t * scale + shift)   (in-place over d_out)
__global__ void k_apply(const float* __restrict__ h, const float* __restrict__ bnscale,
                        const float* __restrict__ bnshift, float* __restrict__ out) {
  int i = blockIdx.x * blockDim.x + threadIdx.x;   // over N*D/4
  if (i >= NNODES * D / 4) return;
  int d4 = (i & (D / 4 - 1)) * 4;
  float4 t  = ((const float4*)out)[i];
  float4 hv = ((const float4*)h)[i];
  float z0v = t.x * bnscale[d4 + 0] + bnshift[d4 + 0];
  float z1v = t.y * bnscale[d4 + 1] + bnshift[d4 + 1];
  float z2v = t.z * bnscale[d4 + 2] + bnshift[d4 + 2];
  float z3v = t.w * bnscale[d4 + 3] + bnshift[d4 + 3];
  float4 r = make_float4(hv.x + (z0v > 0.0f ? z0v : 0.0f),
                         hv.y + (z1v > 0.0f ? z1v : 0.0f),
                         hv.z + (z2v > 0.0f ? z2v : 0.0f),
                         hv.w + (z3v > 0.0f ? z3v : 0.0f));
  ((float4*)out)[i] = r;
}

extern "C" void kernel_launch(void* const* d_in, const int* in_sizes, int n_in,
                              void* d_out, int out_size, void* d_ws, size_t ws_size,
                              hipStream_t stream) {
  const float* h     = (const float*)d_in[0];
  const float* mask  = (const float*)d_in[1];
  const float* snorm = (const float*)d_in[2];
  const float* W1    = (const float*)d_in[3];
  const float* b1    = (const float*)d_in[4];
  const float* W2    = (const float*)d_in[5];
  const float* b2    = (const float*)d_in[6];
  const float* eps   = (const float*)d_in[7];
  const float* gamma = (const float*)d_in[8];
  const float* beta  = (const float*)d_in[9];
  const int*   src   = (const int*)d_in[10];
  const int*   dst   = (const int*)d_in[11];
  float* out = (float*)d_out;

  // workspace layout (all 32B-aligned): z0 f32, y bf16, packed weights, BN accum
  char* w = (char*)d_ws;
  float*  z0  = (float*)(w);                                // 25,600,000 B
  bf16_t* y   = (bf16_t*)(w + (size_t)NNODES * D * 4);      // 12,800,000 B
  bf16_t* w1p = (bf16_t*)(w + (size_t)NNODES * D * 6);      // 32,768 B
  bf16_t* w2p = w1p + 16384;                                // 32,768 B
  float*  gacc = (float*)(w + (size_t)NNODES * D * 6 + 65536);
  float* gsum = gacc, *gsq = gacc + D, *bnscale = gacc + 2 * D, *bnshift = gacc + 3 * D;

  k_prep<<<128, 256, 0, stream>>>(W1, W2, w1p, w2p, gacc);
  k_init<<<(NNODES * D / 4 + 255) / 256, 256, 0, stream>>>(h, eps, z0);
  k_scatter<<<(int)(((long long)NEDGES * 32 + 255) / 256), 256, 0, stream>>>(h, mask, src, dst, z0);
  int gemmBlocks = (NNODES / 16 + 7) / 8;                    // 391
  k_gemm1<<<gemmBlocks, 256, 0, stream>>>(z0, w1p, b1, y);
  k_gemm2<<<gemmBlocks, 256, 0, stream>>>(y, w2p, b2, snorm, out, gsum, gsq);
  k_bnfin<<<1, 128, 0, stream>>>(gsum, gsq, gamma, beta, bnscale, bnshift);
  k_apply<<<(NNODES * D / 4 + 255) / 256, 256, 0, stream>>>(h, bnscale, bnshift, out);
}
